// TransformersBiLstmCrf_33071248180143
// MI455X (gfx1250) — matrix operand order repaired and checked by reading, not compiled, not run on hardware
//
#include <hip/hip_runtime.h>
#include <cstdint>
#include <cstddef>

// ---------------------------------------------------------------------------
// BiLSTM + CRF for MI455X (gfx1250, wave32, WMMA + TDM).
//
// Input GEMMs use v_wmma_f32_16x16x32_bf16 with LDS-staged B tiles. The serial
// time recurrence runs as a persistent 8-workgroup kernel: Whh slab (144KB),
// h state (48KB) and a double-buffered gate tile (2x48KB) live in the 320KB
// WGP LDS; both the Whh slab and the per-step gate tiles are moved by the
// Tensor Data Mover (tensor_load_to_lds, TENSORcnt) so the DMA of step t+1
// overlaps the WMMA compute of step t. Cross-workgroup sync per timestep is
// an atomic phase counter plus the cluster split-barrier.
// ---------------------------------------------------------------------------

typedef __bf16 bf16;
typedef __attribute__((ext_vector_type(16))) __bf16 v16bf;
typedef __attribute__((ext_vector_type(8)))  float  v8f;
typedef __attribute__((ext_vector_type(4)))  unsigned int v4u;
typedef __attribute__((ext_vector_type(8)))  int    v8i;
typedef __attribute__((ext_vector_type(4)))  int    v4i;

#define LSEQ   512
#define NBATCH 64
#define HID    768
#define HALFD  384
#define GATE4  1536
#define NTAG   9
#define NWG    8      // persistent workgroups in the recurrent kernel
#define NC     48     // LSTM cells per workgroup (8*48 = 384)

// Dynamic LDS layout of the recurrent kernel (bytes)
#define WHH_LDS_BYTES (4 * NC * HALFD * 2)              // 147456
#define H_LDS_BYTES   (NBATCH * HALFD * 2)              // 49152
#define GBUF_BYTES    (NBATCH * 4 * NC * 4)             // 49152 (one buffer)
#define REC_SMEM      (WHH_LDS_BYTES + H_LDS_BYTES + 2 * GBUF_BYTES) // 294912

// ---------------------------------------------------------------------------
// Tensor Data Mover: 2D tile load Global -> LDS (D# built per CDNA5 ISA ch.8).
// tile_w elements wide (contiguous, dim0), tile_h rows, row stride in elems.
// dsz_code: 0=1B, 1=2B, 2=4B. Issued once per wave; tracked by TENSORcnt.
// ---------------------------------------------------------------------------
static __device__ __forceinline__ void tdm_load_2d(
    const void* gsrc, unsigned lds_byte_addr, unsigned dsz_code,
    unsigned tile_w, unsigned tile_h, unsigned stride_elems) {
  uint64_t ga = (uint64_t)(uintptr_t)gsrc;
  v4u g0;
  g0[0] = 1u;                                   // count=1, user descriptor
  g0[1] = lds_byte_addr;                        // lds_addr
  g0[2] = (unsigned)ga;                         // global_addr[31:0]
  g0[3] = (unsigned)((ga >> 32) & 0x01FFFFFFu)  // global_addr[56:32]
          | (2u << 30);                         // type=2 ("image")
  v8i g1;
  g1[0] = (int)(dsz_code << 16);                // wg_mask=0 | data_size
  g1[1] = (int)((tile_w & 0xFFFFu) << 16);      // tensor_dim0[15:0]
  g1[2] = (int)(((tile_w >> 16) & 0xFFFFu) | ((tile_h & 0xFFFFu) << 16));
  g1[3] = (int)(((tile_h >> 16) & 0xFFFFu) | ((tile_w & 0xFFFFu) << 16)); // tile_dim0
  g1[4] = (int)(tile_h & 0xFFFFu);              // tile_dim1 | tile_dim2=0
  g1[5] = (int)stride_elems;                    // tensor_dim0_stride[31:0]
  g1[6] = 0;
  g1[7] = 0;
  v4i gz = {0, 0, 0, 0};
#if __clang_major__ >= 23
  v8i gz8 = {0, 0, 0, 0, 0, 0, 0, 0};
  __builtin_amdgcn_tensor_load_to_lds(g0, g1, gz, gz, gz8, 0);
#else
  __builtin_amdgcn_tensor_load_to_lds(g0, g1, gz, gz, 0);
#endif
}

// ---------------------------------------------------------------------------
// WMMA helpers (CDNA5 16x16x32 bf16, f32 accumulate).
// A (16x32, bf16): lane L holds row m = L%16; k-base = (L/16)*8;
//   element e -> k = kbase + (e<8 ? e : e+8)  (two contiguous 16B chunks).
// B (32x16, bf16): lane L holds col n = L%16; k-base = (L/16)*8;
//   element e -> k = kbase + (e<8 ? e : e+8)  (two contiguous 16B chunks).
// C/D (16x16, f32): element r -> (m = (L/16)*8 + r, n = L%16).
// ---------------------------------------------------------------------------
static __device__ __forceinline__ v8f wmma_bf16(v16bf a, v16bf b, v8f c) {
  return __builtin_amdgcn_wmma_f32_16x16x32_bf16(
      /*neg_a=*/false, a, /*neg_b=*/false, b,
      /*c_mod=*/(short)0, c, /*reuse_a=*/false, /*reuse_b=*/false);
}

static __device__ __forceinline__ v16bf load_a_frag(const bf16* base, int ld,
                                                    int row, int kb, int lane) {
  int m  = row + (lane & 15);
  int k0 = kb + ((lane >> 4) << 3);
  const bf16* p = base + (size_t)m * ld + k0;
  union { v16bf v; uint4 q[2]; } u;
  u.q[0] = *(const uint4*)p;          // k0 .. k0+7
  u.q[1] = *(const uint4*)(p + 16);   // k0+16 .. k0+23
  return u.v;
}

static __device__ __forceinline__ v16bf load_b_frag(const bf16* base, int ld,
                                                    int ncol, int kb, int lane) {
  int n  = ncol + (lane & 15);
  int k0 = kb + ((lane >> 4) << 3);
  const bf16* p = base + (size_t)n * ld + k0;
  union { v16bf v; uint4 q[2]; } u;
  u.q[0] = *(const uint4*)p;          // k0 .. k0+7
  u.q[1] = *(const uint4*)(p + 16);   // k0+16 .. k0+23
  return u.v;
}

// ---------------------------------------------------------------------------
// Small conversion kernels
// ---------------------------------------------------------------------------
__global__ void zero_u32_kernel(unsigned* p, int n) {
  int i = blockIdx.x * blockDim.x + threadIdx.x;
  if (i < n) p[i] = 0u;
}

__global__ void cvt_bf16_kernel(const float* __restrict__ in,
                                bf16* __restrict__ out, int n) {
  int i = blockIdx.x * blockDim.x + threadIdx.x;
  int stride = gridDim.x * blockDim.x;
  for (; i < n; i += stride) out[i] = (bf16)in[i];
}

// hidden_states (B,L,768) f32 -> X (L,B,768) bf16 (time-major for recurrence)
__global__ void transpose_x_kernel(const float* __restrict__ hs,
                                   bf16* __restrict__ xo) {
  int row = blockIdx.x;                 // 0 .. L*B-1, row = t*64 + b
  int t = row >> 6, b = row & 63;
  const float* src = hs + ((size_t)b * LSEQ + t) * HID;
  bf16* dst = xo + (size_t)row * HID;
  for (int i = threadIdx.x; i < HID; i += blockDim.x) dst[i] = (bf16)src[i];
}

// ---------------------------------------------------------------------------
// Input-projection GEMM: out(R=L*B, 1536) = X(R,768) @ W(1536,768)^T + bih+bhh
// Block = 8 waves; each wave owns 1 M-tile x 4 N-tiles. B tiles staged in LDS.
// ---------------------------------------------------------------------------
__global__ __launch_bounds__(256)
void gates_gemm_kernel(const bf16* __restrict__ X,
                       const bf16* __restrict__ W,
                       const float* __restrict__ bih,
                       const float* __restrict__ bhh,
                       float* __restrict__ out) {
  __shared__ bf16 bstage[64 * 32];      // [n(64)][k(32)]
  int tid = threadIdx.x, lane = tid & 31, wave = tid >> 5;
  int mt = blockIdx.x * 8 + wave;       // M tile, 0..2047
  int nbase = blockIdx.y * 64;          // 4 N tiles of 16

  v8f acc[4] = {};
  for (int kc = 0; kc < 24; ++kc) {
    __syncthreads();
    {
      int row = tid >> 2;               // 0..63
      int kp  = (tid & 3) * 8;          // 0,8,16,24
      *(uint4*)(bstage + row * 32 + kp) =
          *(const uint4*)(W + (size_t)(nbase + row) * HID + kc * 32 + kp);
    }
    __syncthreads();
    v16bf a = load_a_frag(X, HID, mt * 16, kc * 32, lane);
#pragma unroll
    for (int s = 0; s < 4; ++s) {
      v16bf b = load_b_frag(bstage, 32, s * 16, 0, lane);
      acc[s] = wmma_bf16(a, b, acc[s]);
    }
  }
  int mrow = mt * 16 + ((lane >> 4) << 3);
#pragma unroll
  for (int s = 0; s < 4; ++s) {
    int n = nbase + s * 16 + (lane & 15);
    float bias = bih[n] + bhh[n];
#pragma unroll
    for (int r = 0; r < 8; ++r)
      out[(size_t)(mrow + r) * GATE4 + n] = acc[s][r] + bias;
  }
}

// ---------------------------------------------------------------------------
// Persistent recurrent LSTM kernel: 8 workgroups x 384 threads (12 waves).
// Each WG owns 48 cells. Whh slab + gate tiles arrive via TDM; cell state in
// VGPRs. Per step: 12 waves x 12 K-chunks x 4 gates = 576 WMMAs per WG.
// ---------------------------------------------------------------------------
static __device__ __forceinline__ void grid_barrier(unsigned* cnt, int tid,
                                                    int lane) {
  __threadfence();
  __syncthreads();
  if (tid < 32) {
    // HW cluster split-barrier: real sync if cluster-dispatched, NOP otherwise.
    __builtin_amdgcn_s_cluster_barrier();
    if (lane == 0) {
      unsigned ticket = atomicAdd(cnt, 1u);
      unsigned target = (ticket / NWG + 1u) * NWG;
      while (__hip_atomic_load(cnt, __ATOMIC_ACQUIRE,
                               __HIP_MEMORY_SCOPE_AGENT) < target)
        __builtin_amdgcn_s_sleep(1);
    }
  }
  __syncthreads();
}

__global__ __launch_bounds__(384)
void lstm_rec_kernel(const float* __restrict__ gates,  // (L,B,1536) bias folded
                     const bf16* __restrict__ whh,     // (1536,384)
                     bf16* __restrict__ hx,            // 2 x (64*384) exchange
                     bf16* __restrict__ seq_out,       // (L,B,768)
                     int colOff, int reverse,
                     unsigned* __restrict__ barrier_cnt) {
  extern __shared__ char smem[];
  bf16*  whh_lds = (bf16*)smem;                        // [192][384]
  bf16*  h_lds   = (bf16*)(smem + WHH_LDS_BYTES);      // [64][384]
  float* gbuf    = (float*)(smem + WHH_LDS_BYTES + H_LDS_BYTES); // 2x[4][64][48]

  int wg = blockIdx.x;
  int c0 = wg * NC;
  int tid = threadIdx.x, lane = tid & 31, wave = tid >> 5;  // 12 waves
  int mi = wave / 3, ci = wave % 3;      // 4 M-tiles x 3 cell-tiles
  int m0 = mi * 16, nl0 = ci * 16;

  // Wave-uniform scalar flag: only wave 0 issues TDM ops (TDM ignores EXEC).
  bool firstwave = (__builtin_amdgcn_readfirstlane(tid) < 32);
  unsigned gbuf_lds = (unsigned)(uintptr_t)(void*)gbuf;

  // TDM: pull the Whh slab (4 blocks of 48 contiguous rows x 384 bf16).
  if (firstwave) {
#pragma unroll
    for (int g = 0; g < 4; ++g)
      tdm_load_2d(whh + (size_t)(g * HALFD + c0) * HALFD,
                  (unsigned)(uintptr_t)(void*)whh_lds + g * (NC * HALFD * 2),
                  /*dsz=*/1u, /*tile_w=*/NC * HALFD, /*tile_h=*/1u,
                  /*stride=*/NC * HALFD);
    // TDM: prologue gate tiles for step 0 into buffer 0.
    int tt0 = reverse ? (LSEQ - 1) : 0;
    const float* grow0 = gates + (size_t)tt0 * NBATCH * GATE4;
#pragma unroll
    for (int g = 0; g < 4; ++g)
      tdm_load_2d(grow0 + g * HALFD + c0, gbuf_lds + g * (NBATCH * NC * 4),
                  /*dsz=*/2u, /*tile_w=*/NC, /*tile_h=*/NBATCH,
                  /*stride=*/GATE4);
  }
  // h0 = 0
  for (int i = tid; i < (NBATCH * HALFD) / 8; i += 384)
    ((uint4*)h_lds)[i] = make_uint4(0, 0, 0, 0);
  if (firstwave) __builtin_amdgcn_s_wait_tensorcnt(0);
  __syncthreads();

  float creg[8];
#pragma unroll
  for (int r = 0; r < 8; ++r) creg[r] = 0.f;

  for (int step = 0; step < LSEQ; ++step) {
    int tt = reverse ? (LSEQ - 1 - step) : step;
    int pb = step & 1;

    // TDM: issue step t+1 gate tiles into the other buffer, then wait until
    // only those 4 remain outstanding -> step t's tiles are complete.
    if (firstwave) {
      if (step + 1 < LSEQ) {
        int tn = reverse ? (tt - 1) : (tt + 1);
        const float* grown = gates + (size_t)tn * NBATCH * GATE4;
#pragma unroll
        for (int g = 0; g < 4; ++g)
          tdm_load_2d(grown + g * HALFD + c0,
                      gbuf_lds + (pb ^ 1) * GBUF_BYTES + g * (NBATCH * NC * 4),
                      /*dsz=*/2u, /*tile_w=*/NC, /*tile_h=*/NBATCH,
                      /*stride=*/GATE4);
        __builtin_amdgcn_s_wait_tensorcnt(4);
      } else {
        __builtin_amdgcn_s_wait_tensorcnt(0);
      }
    }
    __syncthreads();
    const float* gcur = gbuf + (size_t)pb * (NBATCH * 4 * NC); // [4][64][48]

    // h_{t-1} @ Whh^T for this WG's 4x(16x16) tile group per wave
    v8f acc[4] = {};
    for (int kc = 0; kc < 12; ++kc) {
      v16bf a = load_a_frag(h_lds, HALFD, m0, kc * 32, lane);
#pragma unroll
      for (int g = 0; g < 4; ++g) {
        v16bf b = load_b_frag(whh_lds, HALFD, g * NC + nl0, kc * 32, lane);
        acc[g] = wmma_bf16(a, b, acc[g]);
      }
    }

    // Activations + state update; write h_t slice to exchange buf + seq out
    int mbase = m0 + ((lane >> 4) << 3);
    int cc = nl0 + (lane & 15);
    bf16* hxw = hx + (size_t)(step & 1) * NBATCH * HALFD;
#pragma unroll
    for (int r = 0; r < 8; ++r) {
      int m = mbase + r;
      float xi = acc[0][r] + gcur[0 * NBATCH * NC + m * NC + cc];
      float xf = acc[1][r] + gcur[1 * NBATCH * NC + m * NC + cc];
      float xg = acc[2][r] + gcur[2 * NBATCH * NC + m * NC + cc];
      float xo = acc[3][r] + gcur[3 * NBATCH * NC + m * NC + cc];
      float si = 1.f / (1.f + __expf(-xi));
      float sf = 1.f / (1.f + __expf(-xf));
      float tg = tanhf(xg);
      float so = 1.f / (1.f + __expf(-xo));
      creg[r] = sf * creg[r] + si * tg;
      float h = so * tanhf(creg[r]);
      bf16 hb = (bf16)h;
      hxw[(size_t)m * HALFD + c0 + cc] = hb;
      seq_out[((size_t)tt * NBATCH + m) * HID + colOff + c0 + cc] = hb;
    }

    // All 8 WGs exchange their h_t slices (double-buffered, 1 barrier/step)
    grid_barrier(barrier_cnt, tid, lane);

    const uint4* src = (const uint4*)(hx + (size_t)(step & 1) * NBATCH * HALFD);
    for (int i = tid; i < (NBATCH * HALFD) / 8; i += 384)
      ((uint4*)h_lds)[i] = src[i];
    __syncthreads();
  }
}

// ---------------------------------------------------------------------------
// Compaction index scan (one serial thread per batch; trivial work)
// ---------------------------------------------------------------------------
__global__ void compact_scan_kernel(const int* __restrict__ valid,
                                    int* __restrict__ srcidx,
                                    int* __restrict__ counts) {
  int b = blockIdx.x;
  if (threadIdx.x == 0) {
    int c = 0;
    for (int t = 0; t < LSEQ; ++t)
      if (valid[b * LSEQ + t]) srcidx[b * LSEQ + c++] = t;
    counts[b] = c;
  }
}

// ---------------------------------------------------------------------------
// Classifier: logits(B,L,9) = compact(seq2) @ cls_w^T + cls_b
// ---------------------------------------------------------------------------
__global__ __launch_bounds__(256)
void logits_kernel(const bf16* __restrict__ seq2,   // (L,B,768)
                   const int* __restrict__ srcidx,
                   const int* __restrict__ counts,
                   const float* __restrict__ cls_w, // (9,768)
                   const float* __restrict__ cls_b,
                   float* __restrict__ logits) {    // (B,L,9)
  int r = blockIdx.x * blockDim.x + threadIdx.x;    // 0..32767
  if (r >= NBATCH * LSEQ) return;
  int b = r >> 9, i = r & 511;
  float acc[NTAG];
#pragma unroll
  for (int t = 0; t < NTAG; ++t) acc[t] = cls_b[t];
  if (i < counts[b]) {
    int t0 = srcidx[b * LSEQ + i];
    const bf16* x = seq2 + ((size_t)t0 * NBATCH + b) * HID;
    for (int k = 0; k < HID; ++k) {
      float xv = (float)x[k];
#pragma unroll
      for (int t = 0; t < NTAG; ++t) acc[t] += xv * cls_w[t * HID + k];
    }
  }
#pragma unroll
  for (int t = 0; t < NTAG; ++t) logits[(size_t)r * NTAG + t] = acc[t];
}

// ---------------------------------------------------------------------------
// Viterbi + CRF log-likelihood. One wave per batch; lanes 0..8 = tags.
// ---------------------------------------------------------------------------
__global__ __launch_bounds__(32)
void crf_kernel(const float* __restrict__ logits,   // (B,L,9)
                const int* __restrict__ labels,
                const int* __restrict__ maskI,      // label_masks
                const float* __restrict__ startw,
                const float* __restrict__ endw,
                const float* __restrict__ trans,
                const int* __restrict__ counts,
                int* __restrict__ hist,             // (B,L,9)
                float* __restrict__ resbuf,         // (B,)
                float* __restrict__ outtags) {      // d_out+1, (B,L) as float
  int b = blockIdx.x, j = threadIdx.x;
  __shared__ float tr[NTAG * NTAG];
  __shared__ float sc[NTAG], al[NTAG], nsc[NTAG], nal[NTAG];
  __shared__ int nptr[NTAG];
  __shared__ int tagtmp[LSEQ];

  for (int i = j; i < NTAG * NTAG; i += 32) tr[i] = trans[i];
  const float* lg = logits + (size_t)b * LSEQ * NTAG;
  if (j < NTAG) { sc[j] = startw[j] + lg[j]; al[j] = sc[j]; }
  __syncthreads();

  for (int i = 1; i < LSEQ; ++i) {
    int m = maskI[b * LSEQ + i];
    if (j < NTAG) {
      float e = lg[i * NTAG + j];
      float best = -1e30f; int bi = 0;
      float mx = -1e30f;
#pragma unroll
      for (int p = 0; p < NTAG; ++p) {
        float v = sc[p] + tr[p * NTAG + j];
        if (v > best) { best = v; bi = p; }
        float w = al[p] + tr[p * NTAG + j];
        if (w > mx) mx = w;
      }
      float sm = 0.f;
#pragma unroll
      for (int p = 0; p < NTAG; ++p)
        sm += __expf(al[p] + tr[p * NTAG + j] - mx);
      nsc[j]  = m ? (best + e) : sc[j];
      nptr[j] = m ? bi : j;
      nal[j]  = m ? (mx + __logf(sm) + e) : al[j];
    }
    __syncthreads();
    if (j < NTAG) {
      sc[j] = nsc[j]; al[j] = nal[j];
      hist[((size_t)b * LSEQ + i) * NTAG + j] = nptr[j];
    }
    __syncthreads();
  }

  if (j == 0) {
    float best = -1e30f; int last = 0;
    for (int p = 0; p < NTAG; ++p) {
      float v = sc[p] + endw[p];
      if (v > best) { best = v; last = p; }
    }
    int tag = last;
    for (int i = LSEQ - 1; i >= 1; --i) {
      tagtmp[i] = tag;
      tag = hist[((size_t)b * LSEQ + i) * NTAG + tag];
    }
    tagtmp[0] = tag;
    for (int i = 0; i < LSEQ; ++i)
      outtags[b * LSEQ + i] = (float)(maskI[b * LSEQ + i] ? tagtmp[i] : 0);

    float mx = -1e30f;
    for (int p = 0; p < NTAG; ++p) {
      float v = al[p] + endw[p];
      if (v > mx) mx = v;
    }
    float sm = 0.f;
    for (int p = 0; p < NTAG; ++p) sm += __expf(al[p] + endw[p] - mx);
    float den = mx + __logf(sm);

    const int* lab = labels + b * LSEQ;
    float num = startw[lab[0]] + lg[lab[0]];
    for (int i = 1; i < LSEQ; ++i)
      if (maskI[b * LSEQ + i])
        num += tr[lab[i - 1] * NTAG + lab[i]] + lg[i * NTAG + lab[i]];
    num += endw[lab[counts[b] - 1]];
    resbuf[b] = num - den;
  }
}

__global__ void final_reduce_kernel(const float* __restrict__ resbuf,
                                    float* __restrict__ out0) {
  if (threadIdx.x == 0) {
    float s = 0.f;
    for (int b = 0; b < NBATCH; ++b) s += resbuf[b];
    out0[0] = -s;
  }
}

// ---------------------------------------------------------------------------
// Host orchestration
// ---------------------------------------------------------------------------
extern "C" void kernel_launch(void* const* d_in, const int* in_sizes, int n_in,
                              void* d_out, int out_size, void* d_ws,
                              size_t ws_size, hipStream_t stream) {
  (void)in_sizes; (void)n_in; (void)out_size; (void)ws_size;
  const float* hs    = (const float*)d_in[0];
  const float* w_ih  = (const float*)d_in[1];   // (2,2,1536,768)
  const float* w_hh  = (const float*)d_in[2];   // (2,2,1536,384)
  const float* b_ih  = (const float*)d_in[3];   // (2,2,1536)
  const float* b_hh  = (const float*)d_in[4];
  const float* cls_w = (const float*)d_in[5];
  const float* cls_b = (const float*)d_in[6];
  const float* startw= (const float*)d_in[7];
  const float* endw  = (const float*)d_in[8];
  const float* trans = (const float*)d_in[9];
  const int*   valid = (const int*)d_in[10];
  const int*   labels= (const int*)d_in[11];
  const int*   lmask = (const int*)d_in[12];
  float* out = (float*)d_out;

  char* base = (char*)d_ws;
  size_t off = 0;
  auto carve = [&](size_t bytes) -> char* {
    char* p = base + off;
    off = (off + bytes + 255) & ~(size_t)255;
    return p;
  };
  const size_t R = (size_t)LSEQ * NBATCH;                 // 32768 rows
  bf16*  X16    = (bf16*)carve(R * HID * 2);
  bf16*  WIH16  = (bf16*)carve(4ull * GATE4 * HID * 2);
  bf16*  WHH16  = (bf16*)carve(4ull * GATE4 * HALFD * 2);
  float* GATES0 = (float*)carve(R * GATE4 * 4);
  float* GATES1 = (float*)carve(R * GATE4 * 4);
  bf16*  SEQ1   = (bf16*)carve(R * HID * 2);
  bf16*  SEQ2   = (bf16*)carve(R * HID * 2);
  bf16*  HX     = (bf16*)carve(2ull * NBATCH * HALFD * 2);
  unsigned* CNT = (unsigned*)carve(16 * sizeof(unsigned));
  int*   SRC    = (int*)carve(R * 4);
  int*   COUNTS = (int*)carve(256);
  float* LOGITS = (float*)carve(R * NTAG * 4);
  int*   HIST   = (int*)carve(R * NTAG * 4);
  float* RES    = (float*)carve(256);

  hipFuncSetAttribute((const void*)lstm_rec_kernel,
                      hipFuncAttributeMaxDynamicSharedMemorySize, REC_SMEM);

  zero_u32_kernel<<<1, 64, 0, stream>>>(CNT, 16);

  transpose_x_kernel<<<dim3((unsigned)R), 256, 0, stream>>>(hs, X16);
  cvt_bf16_kernel<<<4096, 256, 0, stream>>>(w_ih, WIH16, 4 * GATE4 * HID);
  cvt_bf16_kernel<<<4096, 256, 0, stream>>>(w_hh, WHH16, 4 * GATE4 * HALFD);

  const bf16* layer_in = X16;
  bf16* layer_out[2] = {SEQ1, SEQ2};
  for (int l = 0; l < 2; ++l) {
    for (int d = 0; d < 2; ++d) {
      int dl = l * 2 + d;
      float* gates = (d == 0) ? GATES0 : GATES1;
      gates_gemm_kernel<<<dim3(256, 24), 256, 0, stream>>>(
          layer_in, WIH16 + (size_t)dl * GATE4 * HID,
          b_ih + (size_t)dl * GATE4, b_hh + (size_t)dl * GATE4, gates);
    }
    for (int d = 0; d < 2; ++d) {
      int dl = l * 2 + d;
      float* gates = (d == 0) ? GATES0 : GATES1;
      lstm_rec_kernel<<<NWG, 384, REC_SMEM, stream>>>(
          gates, WHH16 + (size_t)dl * GATE4 * HALFD, HX, layer_out[l],
          /*colOff=*/d * HALFD, /*reverse=*/d, CNT + dl);
    }
    layer_in = layer_out[l];
  }

  compact_scan_kernel<<<NBATCH, 32, 0, stream>>>(valid, SRC, COUNTS);
  logits_kernel<<<(NBATCH * LSEQ) / 256, 256, 0, stream>>>(
      SEQ2, SRC, COUNTS, cls_w, cls_b, LOGITS);
  crf_kernel<<<NBATCH, 32, 0, stream>>>(LOGITS, labels, lmask, startw, endw,
                                        trans, COUNTS, HIST, RES, out + 1);
  final_reduce_kernel<<<1, 32, 0, stream>>>(RES, out);
}
